// DecoderWithAttention_91311004713349
// MI455X (gfx1250) — matrix operand order repaired and checked by reading, not compiled, not run on hardware
//
#include <hip/hip_runtime.h>
#include <hip/hip_bf16.h>

// ---------------- problem constants ----------------
static constexpr int kB    = 128;
static constexpr int kP    = 196;    // 14*14 pixels
static constexpr int kE    = 2048;   // encoder dim
static constexpr int kA    = 512;    // attention dim
static constexpr int kEMB  = 512;
static constexpr int kD    = 512;    // decoder dim
static constexpr int kV    = 10000;
static constexpr int kLC   = 128;
static constexpr int kTCAP = 26;
static constexpr int kT    = 25;     // TCAP-1 time steps
static constexpr int kXK   = kEMB + kLC + kE + kD; // 3200 (x' = [emb,style,gate*awe,h])
static constexpr int kG4   = 4 * kD; // 2048 gates

// ---------------- vector types for WMMA ----------------
typedef __bf16 bf16_t;
typedef __attribute__((ext_vector_type(16))) __bf16 bf16x16;
typedef __attribute__((ext_vector_type(8)))  __bf16 bf16x8;
typedef __attribute__((ext_vector_type(8)))  float  floatx8;

__device__ inline float sigf(float x)  { return 1.0f / (1.0f + __expf(-x)); }

// Build 16x32-bf16 A fragment for this lane.
// Lane layout (ISA 7.12.2): row = lane%16.., hf = lane>>4:
//   VGPR0..3 : K = hf*8 + 0..7 ;  VGPR4..7 : K = 16 + hf*8 + 0..7
// p already points at A[row] + k + hf*8.
__device__ inline bf16x16 load_a_frag(const bf16_t* p) {
    bf16x8 lo = *(const bf16x8*)(p);
    bf16x8 hi = *(const bf16x8*)(p + 16);
    bf16x16 a;
#pragma unroll
    for (int i = 0; i < 8; ++i) { a[i] = lo[i]; a[8 + i] = hi[i]; }
    return a;
}

// ---------------- WMMA GEMM: C(MxN,f32) = A(MxK,bf16,row) * Bt(NxK,bf16,row)^T + bias ----
// Per wave: 64(M) x 16(N) tile, 4 accumulators.  Requires M%64==0, N%16==0, K%32==0.
template <bool MASKED>
__global__ __launch_bounds__(256)
void gemm_bf16_wmma(const bf16_t* __restrict__ A, const bf16_t* __restrict__ Bt,
                    const float* __restrict__ bias, float* __restrict__ C,
                    int M, int N, int K, long ldC,
                    const int* __restrict__ rowlen, int tstep)
{
    const int lane = threadIdx.x & 31;
    const int wid  = threadIdx.x >> 5;
    const int n0   = blockIdx.y * 128 + wid * 16;
    if (n0 >= N) return;                    // wave-uniform skip (EXEC stays all-ones)
    const int m0   = blockIdx.x * 64;
    const int col  = lane & 15;
    const int hf   = lane >> 4;

    floatx8 acc0 = {}, acc1 = {}, acc2 = {}, acc3 = {};

    // B fragment: lane -> column n0+col, K = k + hf*8 + 0..7 and k + 16 + hf*8 + 0..7 (same as A).
    const bf16_t* brow  = Bt + (long)(n0 + col) * K + hf * 8;
    const bf16_t* arow0 = A + (long)(m0 +  0 + col) * K + hf * 8;
    const bf16_t* arow1 = A + (long)(m0 + 16 + col) * K + hf * 8;
    const bf16_t* arow2 = A + (long)(m0 + 32 + col) * K + hf * 8;
    const bf16_t* arow3 = A + (long)(m0 + 48 + col) * K + hf * 8;

    for (int k = 0; k < K; k += 32) {
        bf16x16 bf = load_a_frag(brow + k);
        bf16x16 a0 = load_a_frag(arow0 + k);
        bf16x16 a1 = load_a_frag(arow1 + k);
        bf16x16 a2 = load_a_frag(arow2 + k);
        bf16x16 a3 = load_a_frag(arow3 + k);
        acc0 = __builtin_amdgcn_wmma_f32_16x16x32_bf16(false, a0, false, bf, (short)0, acc0, false, false);
        acc1 = __builtin_amdgcn_wmma_f32_16x16x32_bf16(false, a1, false, bf, (short)0, acc1, false, false);
        acc2 = __builtin_amdgcn_wmma_f32_16x16x32_bf16(false, a2, false, bf, (short)0, acc2, false, false);
        acc3 = __builtin_amdgcn_wmma_f32_16x16x32_bf16(false, a3, false, bf, (short)0, acc3, false, false);
    }

    // Epilogue. D layout: VGPR r holds row (r + 8*hf), column = col.
    const int n  = n0 + col;
    const float bv = bias[n];
#pragma unroll
    for (int r = 0; r < 8; ++r) {
        const int mr = hf * 8 + r;
        float v0 = acc0[r] + bv, v1 = acc1[r] + bv, v2 = acc2[r] + bv, v3 = acc3[r] + bv;
        if (MASKED) {
            v0 = (tstep < rowlen[m0 +  0 + mr]) ? v0 : 0.0f;
            v1 = (tstep < rowlen[m0 + 16 + mr]) ? v1 : 0.0f;
            v2 = (tstep < rowlen[m0 + 32 + mr]) ? v2 : 0.0f;
            v3 = (tstep < rowlen[m0 + 48 + mr]) ? v3 : 0.0f;
        }
        C[(long)(m0 +  0 + mr) * ldC + n] = v0;
        C[(long)(m0 + 16 + mr) * ldC + n] = v1;
        C[(long)(m0 + 32 + mr) * ldC + n] = v2;
        C[(long)(m0 + 48 + mr) * ldC + n] = v3;
    }
}

// ---------------- phase-0 helper kernels ----------------

// argsort(-cap_len) stable, gather caps/lengths/classes, emit int + float copies.
__global__ void sort_gather_kernel(const int* __restrict__ cap_len, const int* __restrict__ caps,
                                   const int* __restrict__ length_class,
                                   int* sidx, int* dlen, int* cls, int* caps_s,
                                   float* out_caps, float* out_dlen, float* out_sidx)
{
    __shared__ int len[kB];
    __shared__ int perm[kB];
    const int i = threadIdx.x;
    len[i] = cap_len[i];            // caption_lengths is (B,1)
    __syncthreads();
    const int li = len[i];
    int rank = 0;
    for (int j = 0; j < kB; ++j) {
        const int lj = len[j];
        rank += (lj > li) || (lj == li && j < i);
    }
    perm[rank] = i;
    __syncthreads();
    const int src = perm[i];
    sidx[i] = src;            out_sidx[i] = (float)src;
    const int dl = len[src] - 1;
    dlen[i] = dl;             out_dlen[i] = (float)dl;
    cls[i]  = length_class[src];
    for (int j = 0; j < kTCAP; ++j) {
        const int cv = caps[src * kTCAP + j];
        caps_s[i * kTCAP + j]   = cv;
        out_caps[i * kTCAP + j] = (float)cv;
    }
}

// dst(N x K, bf16) = transpose of src(K x N, f32)
__global__ void convt_kernel(const float* __restrict__ src, bf16_t* __restrict__ dst,
                             int K, int N)
{
    const long idx = (long)blockIdx.x * 256 + threadIdx.x;
    if (idx >= (long)K * N) return;
    const int n = (int)(idx / K);
    const int k = (int)(idx % K);
    dst[idx] = (bf16_t)src[(long)k * N + n];
}

// Combined [W_ih ; W_hh] transposed into (2048 x 3200) bf16.
__global__ void comb_kernel(const float* __restrict__ W_ih, const float* __restrict__ W_hh,
                            bf16_t* __restrict__ dst)
{
    const long idx = (long)blockIdx.x * 256 + threadIdx.x;
    if (idx >= (long)kG4 * kXK) return;
    const int n = (int)(idx / kXK);
    const int k = (int)(idx % kXK);
    const float v = (k < kEMB + kLC + kE) ? W_ih[(long)k * kG4 + n]
                                          : W_hh[(long)(k - (kEMB + kLC + kE)) * kG4 + n];
    dst[idx] = (bf16_t)v;
}

__global__ void bias_comb_kernel(const float* __restrict__ b_ih, const float* __restrict__ b_hh,
                                 float* __restrict__ bc)
{
    const int i = blockIdx.x * 256 + threadIdx.x;
    if (i < kG4) bc[i] = b_ih[i] + b_hh[i];
}

// Gather+convert encoder rows by sort index into bf16 (B,196,2048).
__global__ __launch_bounds__(256)
void enc_gather_kernel(const float* __restrict__ enc, const int* __restrict__ sidx,
                       bf16_t* __restrict__ encb)
{
    const int b = blockIdx.x, p = blockIdx.y;
    const float*  s = enc  + ((long)sidx[b] * kP + p) * kE;
    bf16_t*       d = encb + ((long)b       * kP + p) * kE;
    for (int e = threadIdx.x; e < kE; e += 256) d[e] = (bf16_t)s[e];
}

// mean over pixels (sorted order), fp32 + bf16.
__global__ __launch_bounds__(256)
void mean_kernel(const float* __restrict__ enc, const int* __restrict__ sidx,
                 float* __restrict__ meanf, bf16_t* __restrict__ meanb)
{
    const int b = blockIdx.x;
    const int e = blockIdx.y * 256 + threadIdx.x;
    const float* s = enc + (long)sidx[b] * kP * kE + e;
    float acc = 0.0f;
    for (int p = 0; p < kP; ++p) acc += s[(long)p * kE];
    acc *= (1.0f / (float)kP);
    meanf[b * kE + e] = acc;
    meanb[b * kE + e] = (bf16_t)acc;
}

__global__ void f2bf_kernel(const float* __restrict__ src, bf16_t* __restrict__ dst, int n)
{
    const int i = blockIdx.x * 256 + threadIdx.x;
    if (i < n) dst[i] = (bf16_t)src[i];
}

// ---------------- per-step kernels ----------------

// e = relu(att1 + att2) . w_full ; alpha = softmax_p(e); masked alphas output.
__global__ __launch_bounds__(256)
void attn_kernel(const float* __restrict__ att1, const float* __restrict__ att2,
                 const float* __restrict__ w_full, const int* __restrict__ dlen, int tstep,
                 float* __restrict__ alpha, float* __restrict__ out_alphas)
{
    const int b = blockIdx.x;
    __shared__ float s2[kA];
    __shared__ float se[kP];
    __shared__ float red[256];
    for (int a = threadIdx.x; a < kA; a += 256) s2[a] = att2[b * kA + a];
    __syncthreads();
    const int wid = threadIdx.x >> 5, lane = threadIdx.x & 31;
    for (int p = wid; p < kP; p += 8) {
        const float* row = att1 + ((long)b * kP + p) * kA;
        float acc = 0.0f;
        for (int a = lane; a < kA; a += 32) {
            float v = row[a] + s2[a];
            v = v > 0.0f ? v : 0.0f;
            acc += v * w_full[a];
        }
        for (int off = 16; off > 0; off >>= 1) acc += __shfl_xor(acc, off, 32);
        if (lane == 0) se[p] = acc;
    }
    __syncthreads();
    const float ev = (threadIdx.x < kP) ? se[threadIdx.x] : -3.4e38f;
    red[threadIdx.x] = ev;
    __syncthreads();
    for (int s = 128; s > 0; s >>= 1) {
        if (threadIdx.x < s) red[threadIdx.x] = fmaxf(red[threadIdx.x], red[threadIdx.x + s]);
        __syncthreads();
    }
    const float mx = red[0];
    __syncthreads();
    const float ex = (threadIdx.x < kP) ? __expf(ev - mx) : 0.0f;
    red[threadIdx.x] = ex;
    __syncthreads();
    for (int s = 128; s > 0; s >>= 1) {
        if (threadIdx.x < s) red[threadIdx.x] += red[threadIdx.x + s];
        __syncthreads();
    }
    const float inv = 1.0f / red[0];
    if (threadIdx.x < kP) {
        const float al = ex * inv;
        alpha[b * kP + threadIdx.x] = al;
        out_alphas[((long)b * kT + tstep) * kP + threadIdx.x] = (tstep < dlen[b]) ? al : 0.0f;
    }
}

// awe[b][e] = sum_p alpha[b][p] * encb[b][p][e]   (bf16 stream, fp32 accumulate)
__global__ __launch_bounds__(256)
void awe_kernel(const bf16_t* __restrict__ encb, const float* __restrict__ alpha,
                float* __restrict__ awe)
{
    const int b = blockIdx.x;
    const int e = blockIdx.y * 256 + threadIdx.x;
    __shared__ float sal[kP];
    if (threadIdx.x < kP) sal[threadIdx.x] = alpha[b * kP + threadIdx.x];
    __syncthreads();
    const bf16_t* s = encb + (long)b * kP * kE + e;
    float acc = 0.0f;
    for (int p = 0; p < kP; ++p) {
        __builtin_prefetch(s + (long)(p + 4) * kE, 0, 1);
        acc += sal[p] * (float)s[(long)p * kE];
    }
    awe[b * kE + e] = acc;
}

// x' = bf16([emb_t, style, sigmoid(gate_pre)*awe, h])  -- 3200 per row
__global__ __launch_bounds__(256)
void xbuild_kernel(const float* __restrict__ emb_table, const int* __restrict__ caps_s, int tstep,
                   const float* __restrict__ lc_table, const int* __restrict__ cls,
                   const float* __restrict__ gate_pre, const float* __restrict__ awe,
                   const float* __restrict__ h, bf16_t* __restrict__ x)
{
    const int b = blockIdx.x;
    const int k = blockIdx.y * 256 + threadIdx.x;
    if (k >= kXK) return;
    float v;
    if (k < kEMB) {
        v = emb_table[(long)caps_s[b * kTCAP + tstep] * kEMB + k];
    } else if (k < kEMB + kLC) {
        v = lc_table[cls[b] * kLC + (k - kEMB)];
    } else if (k < kEMB + kLC + kE) {
        const int j = k - (kEMB + kLC);
        v = sigf(gate_pre[b * kE + j]) * awe[b * kE + j];
    } else {
        v = h[b * kD + (k - (kEMB + kLC + kE))];
    }
    x[(long)b * kXK + k] = (bf16_t)v;
}

// LSTM pointwise with masked state update; keeps h_bf in sync.
__global__ __launch_bounds__(256)
void lstm_kernel(const float* __restrict__ gates, const int* __restrict__ dlen, int tstep,
                 float* __restrict__ h, float* __restrict__ c, bf16_t* __restrict__ h_bf)
{
    const int b = blockIdx.x;
    const int d = blockIdx.y * 256 + threadIdx.x;
    if (d >= kD) return;
    const float* g = gates + (long)b * kG4;
    const float ig = sigf(g[d]);
    const float fg = sigf(g[kD + d]);
    const float gg = tanhf(g[2 * kD + d]);
    const float og = sigf(g[3 * kD + d]);
    if (tstep < dlen[b]) {
        const float cn = fg * c[b * kD + d] + ig * gg;
        const float hn = og * tanhf(cn);
        c[b * kD + d] = cn;
        h[b * kD + d] = hn;
        h_bf[b * kD + d] = (bf16_t)hn;
    }
}

// ---------------- host side ----------------
static inline void launch_gemm(const bf16_t* A, const bf16_t* Bt, const float* bias, float* C,
                               int M, int N, int K, long ldC,
                               const int* rowlen, int tstep, hipStream_t stream)
{
    dim3 grid(M / 64, (N + 127) / 128);
    if (rowlen)
        gemm_bf16_wmma<true><<<grid, 256, 0, stream>>>(A, Bt, bias, C, M, N, K, ldC, rowlen, tstep);
    else
        gemm_bf16_wmma<false><<<grid, 256, 0, stream>>>(A, Bt, bias, C, M, N, K, ldC, nullptr, 0);
}

extern "C" void kernel_launch(void* const* d_in, const int* in_sizes, int n_in,
                              void* d_out, int out_size, void* d_ws, size_t ws_size,
                              hipStream_t stream)
{
    (void)in_sizes; (void)n_in; (void)out_size; (void)ws_size;
    // ---- inputs (setup_inputs order) ----
    const float* encoder_out  = (const float*)d_in[0];
    const int*   enc_caps     = (const int*)  d_in[1];
    const int*   cap_len      = (const int*)  d_in[2];
    const int*   length_class = (const int*)  d_in[3];
    const float* W_enc_att    = (const float*)d_in[4];
    const float* b_enc_att    = (const float*)d_in[5];
    const float* W_dec_att    = (const float*)d_in[6];
    const float* b_dec_att    = (const float*)d_in[7];
    const float* w_full_att   = (const float*)d_in[8];
    /* b_full_att (d_in[9]) is a scalar shift: softmax-invariant, skipped */
    const float* W_init_h     = (const float*)d_in[10];
    const float* b_init_h     = (const float*)d_in[11];
    const float* W_init_c     = (const float*)d_in[12];
    const float* b_init_c     = (const float*)d_in[13];
    const float* W_fbeta      = (const float*)d_in[14];
    const float* b_fbeta      = (const float*)d_in[15];
    const float* emb_table    = (const float*)d_in[16];
    const float* lc_table     = (const float*)d_in[17];
    const float* W_fc         = (const float*)d_in[18];
    const float* b_fc         = (const float*)d_in[19];
    const float* W_ih         = (const float*)d_in[20];
    const float* b_ih         = (const float*)d_in[21];
    const float* W_hh         = (const float*)d_in[22];
    const float* b_hh         = (const float*)d_in[23];

    // ---- output layout (flat, return order) ----
    float* out       = (float*)d_out;
    float* out_pred  = out;                                   // B*T*V
    float* out_caps  = out_pred + (long)kB * kT * kV;         // B*TCAP
    float* out_dlen  = out_caps + (long)kB * kTCAP;           // B
    float* out_alph  = out_dlen + kB;                         // B*T*196
    float* out_sidx  = out_alph + (long)kB * kT * kP;         // B

    // ---- workspace bump allocator (256B aligned) ----
    char* wp = (char*)d_ws;
    auto alloc = [&](size_t bytes) -> void* {
        void* r = (void*)wp;
        wp += (bytes + 255) & ~(size_t)255;
        return r;
    };
    bf16_t* encb    = (bf16_t*)alloc((size_t)kB * kP * kE * 2);
    float*  att1    = (float*) alloc((size_t)kB * kP * kA * 4);
    bf16_t* WfcT    = (bf16_t*)alloc((size_t)kV * kD * 2);
    bf16_t* WcombT  = (bf16_t*)alloc((size_t)kG4 * kXK * 2);
    bf16_t* WencT   = (bf16_t*)alloc((size_t)kA * kE * 2);
    bf16_t* WdecT   = (bf16_t*)alloc((size_t)kA * kD * 2);
    bf16_t* WfbT    = (bf16_t*)alloc((size_t)kE * kD * 2);
    bf16_t* WinitHT = (bf16_t*)alloc((size_t)kD * kE * 2);
    bf16_t* WinitCT = (bf16_t*)alloc((size_t)kD * kE * 2);
    float*  meanf   = (float*) alloc((size_t)kB * kE * 4);
    bf16_t* meanb   = (bf16_t*)alloc((size_t)kB * kE * 2);
    float*  h       = (float*) alloc((size_t)kB * kD * 4);
    float*  c       = (float*) alloc((size_t)kB * kD * 4);
    bf16_t* h_bf    = (bf16_t*)alloc((size_t)kB * kD * 2);
    float*  att2    = (float*) alloc((size_t)kB * kA * 4);
    float*  alpha   = (float*) alloc((size_t)kB * kP * 4);
    float*  awe     = (float*) alloc((size_t)kB * kE * 4);
    float*  gatep   = (float*) alloc((size_t)kB * kE * 4);
    bf16_t* xbuf    = (bf16_t*)alloc((size_t)kB * kXK * 2);
    float*  gates   = (float*) alloc((size_t)kB * kG4 * 4);
    float*  biasc   = (float*) alloc((size_t)kG4 * 4);
    int*    sidx    = (int*)   alloc((size_t)kB * 4);
    int*    dlen    = (int*)   alloc((size_t)kB * 4);
    int*    cls     = (int*)   alloc((size_t)kB * 4);
    int*    caps_s  = (int*)   alloc((size_t)kB * kTCAP * 4);

    // ---- phase 0: sort / convert / precompute ----
    sort_gather_kernel<<<1, kB, 0, stream>>>(cap_len, enc_caps, length_class,
                                             sidx, dlen, cls, caps_s,
                                             out_caps, out_dlen, out_sidx);

    auto convt = [&](const float* src, bf16_t* dst, int K, int N) {
        long total = (long)K * N;
        convt_kernel<<<(unsigned)((total + 255) / 256), 256, 0, stream>>>(src, dst, K, N);
    };
    convt(W_enc_att, WencT,   kE, kA);   // (2048x512)  -> 512x2048
    convt(W_dec_att, WdecT,   kD, kA);   // (512x512)   -> 512x512
    convt(W_fbeta,   WfbT,    kD, kE);   // (512x2048)  -> 2048x512
    convt(W_fc,      WfcT,    kD, kV);   // (512x10000) -> 10000x512
    convt(W_init_h,  WinitHT, kE, kD);   // (2048x512)  -> 512x2048
    convt(W_init_c,  WinitCT, kE, kD);

    {
        long total = (long)kG4 * kXK;
        comb_kernel<<<(unsigned)((total + 255) / 256), 256, 0, stream>>>(W_ih, W_hh, WcombT);
        bias_comb_kernel<<<(kG4 + 255) / 256, 256, 0, stream>>>(b_ih, b_hh, biasc);
    }

    enc_gather_kernel<<<dim3(kB, kP), 256, 0, stream>>>(encoder_out, sidx, encb);
    mean_kernel<<<dim3(kB, kE / 256), 256, 0, stream>>>(encoder_out, sidx, meanf, meanb);

    // h0, c0
    launch_gemm(meanb, WinitHT, b_init_h, h, kB, kD, kE, kD, nullptr, 0, stream);
    launch_gemm(meanb, WinitCT, b_init_c, c, kB, kD, kE, kD, nullptr, 0, stream);
    f2bf_kernel<<<(kB * kD + 255) / 256, 256, 0, stream>>>(h, h_bf, kB * kD);

    // att1 = enc @ W_enc_att + b   (the big GEMM: 25088 x 512 x 2048)
    launch_gemm(encb, WencT, b_enc_att, att1, kB * kP, kA, kE, kA, nullptr, 0, stream);

    // ---- sequential decode ----
    for (int t = 0; t < kT; ++t) {
        // att2 = h @ W_dec_att + b
        launch_gemm(h_bf, WdecT, b_dec_att, att2, kB, kA, kD, kA, nullptr, 0, stream);
        // attention scores + softmax (+ masked alphas out)
        attn_kernel<<<kB, 256, 0, stream>>>(att1, att2, w_full_att, dlen, t, alpha, out_alph);
        // attention-weighted encoding
        awe_kernel<<<dim3(kB, kE / 256), 256, 0, stream>>>(encb, alpha, awe);
        // gate preact = h @ W_fbeta + b
        launch_gemm(h_bf, WfbT, b_fbeta, gatep, kB, kE, kD, kE, nullptr, 0, stream);
        // x' = [emb_t, style, sigmoid(gate)*awe, h]
        xbuild_kernel<<<dim3(kB, (kXK + 255) / 256), 256, 0, stream>>>(
            emb_table, caps_s, t, lc_table, cls, gatep, awe, h, xbuf);
        // gates = x' @ [W_ih;W_hh] + (b_ih + b_hh)
        launch_gemm(xbuf, WcombT, biasc, gates, kB, kG4, kXK, kG4, nullptr, 0, stream);
        // LSTM pointwise, masked state update
        lstm_kernel<<<dim3(kB, (kD + 255) / 256), 256, 0, stream>>>(gates, dlen, t, h, c, h_bf);
        // preds[:, t, :] = h @ W_fc + b_fc (masked to 0 where t >= dec_len)
        launch_gemm(h_bf, WfcT, b_fc, out_pred + (long)t * kV,
                    kB, kV, kD, (long)kT * kV, dlen, t, stream);
    }
}
